// BasicTransformerBlock_11519102288128
// MI455X (gfx1250) — matrix operand order repaired and checked by reading, not compiled, not run on hardware
//
#include <hip/hip_runtime.h>
#include <cstdint>
#include <cstddef>

#define B_     2
#define N_     2048
#define D_     1024
#define CTX_   256
#define TV_    250
#define H_     16
#define DH_    64
#define INNER_ 1024
#define FF_    4096

typedef __attribute__((ext_vector_type(16))) __bf16 v16bf;
typedef __attribute__((ext_vector_type(8)))  float  v8f;

union Frag16 {
    v16bf v;
    uint4 u[2];
    unsigned short s[16];
};

__device__ __forceinline__ unsigned short f2bf(float f) {
    union { float f; unsigned int u; } cv; cv.f = f;
    unsigned int u = cv.u;
    u += 0x7FFFu + ((u >> 16) & 1u);
    return (unsigned short)(u >> 16);
}

// gfx1250 async global->LDS copy (ASYNCcnt-tracked, bypasses VGPRs).
__device__ __forceinline__ void async_ld_b128(unsigned lds_off, const void* gaddr) {
    asm volatile("global_load_async_to_lds_b128 %0, %1, off"
                 :: "v"(lds_off), "v"(gaddr)
                 : "memory");
}
// Async loads complete in order: waiting ASYNCcnt<=N with N newer loads in
// flight guarantees all older loads have landed in LDS.
__device__ __forceinline__ void wait_async0() { asm volatile("s_wait_asynccnt 0x0" ::: "memory"); }
__device__ __forceinline__ void wait_async2() { asm volatile("s_wait_asynccnt 0x2" ::: "memory"); }
__device__ __forceinline__ void wait_async4() { asm volatile("s_wait_asynccnt 0x4" ::: "memory"); }
__device__ __forceinline__ unsigned lds_off32(const void* p) {
    return (unsigned)(size_t)p;   // LDS generic ptr: high 32 = aperture, low 32 = LDS addr
}

// ---------------- elementwise f32 -> bf16 ----------------
__global__ void __launch_bounds__(256) k_cvt(const float* __restrict__ in,
                                             unsigned short* __restrict__ out, size_t n) {
    size_t i  = (size_t)blockIdx.x * blockDim.x + threadIdx.x;
    size_t st = (size_t)gridDim.x * blockDim.x;
    for (; i < n; i += st) out[i] = f2bf(in[i]);
}

// ---------------- transposed convert: in (R x C) f32 -> out (C x R) bf16 -------
__global__ void __launch_bounds__(256) k_cvt_t(const float* __restrict__ in,
                                               unsigned short* __restrict__ out,
                                               int R, int C) {
    size_t n  = (size_t)R * C;
    size_t i  = (size_t)blockIdx.x * blockDim.x + threadIdx.x;
    size_t st = (size_t)gridDim.x * blockDim.x;
    for (; i < n; i += st) {
        int r = (int)(i / C);
        int c = (int)(i % C);
        out[(size_t)c * R + r] = f2bf(in[i]);
    }
}

// ---------------- hint + sinusoidal PE -> bf16 ----------------
__global__ void __launch_bounds__(256) k_hint_pe(const float* __restrict__ hint,
                                                 unsigned short* __restrict__ out,
                                                 int T, int D, size_t n) {
    size_t i  = (size_t)blockIdx.x * blockDim.x + threadIdx.x;
    size_t st = (size_t)gridDim.x * blockDim.x;
    for (; i < n; i += st) {
        int t = (int)((i / D) % T);
        int d = (int)(i % D);
        float i2  = (float)(d >> 1);
        float den = __powf(10000.f, 2.f * i2 / (float)D);
        float ang = (float)t / den;
        float pe  = (d & 1) ? __cosf(ang) : __sinf(ang);
        out[i] = f2bf(hint[i] + pe);
    }
}

// ---------------- LayerNorm (row of D=1024) -> bf16 ----------------
__global__ void __launch_bounds__(256) k_ln(const float* __restrict__ x,
                                            const float* __restrict__ g,
                                            const float* __restrict__ b,
                                            unsigned short* __restrict__ out, int D) {
    __shared__ float red[256];
    const int tid = threadIdx.x;
    const int row = blockIdx.x;
    const float* xr = x + (size_t)row * D;
    float xv[4];
    float s = 0.f;
    for (int j = 0; j < 4; ++j) { xv[j] = xr[tid + j * 256]; s += xv[j]; }
    red[tid] = s; __syncthreads();
    for (int st = 128; st > 0; st >>= 1) { if (tid < st) red[tid] += red[tid + st]; __syncthreads(); }
    float mu = red[0] / (float)D;
    __syncthreads();
    s = 0.f;
    for (int j = 0; j < 4; ++j) { float d = xv[j] - mu; s += d * d; }
    red[tid] = s; __syncthreads();
    for (int st = 128; st > 0; st >>= 1) { if (tid < st) red[tid] += red[tid + st]; __syncthreads(); }
    float rstd = rsqrtf(red[0] / (float)D + 1e-5f);
    for (int j = 0; j < 4; ++j) {
        int i = tid + j * 256;
        out[(size_t)row * D + i] = f2bf((xv[j] - mu) * rstd * g[i] + b[i]);
    }
}

// ---------------- GEGLU: a * gelu_exact(gate) -> bf16 ----------------
__global__ void __launch_bounds__(256) k_geglu(const float* __restrict__ a,
                                               const float* __restrict__ gt,
                                               unsigned short* __restrict__ out, size_t n) {
    size_t i  = (size_t)blockIdx.x * blockDim.x + threadIdx.x;
    size_t st = (size_t)gridDim.x * blockDim.x;
    for (; i < n; i += st) {
        float g = gt[i];
        float ge = 0.5f * g * (1.f + erff(g * 0.70710678118654752f));
        out[i] = f2bf(a[i] * ge);
    }
}

// ---- bf16 WMMA GEMM: C(MxN) = A(MxK,row) * Bt(NxK,row == B^T) [+bias][+resid] ----
// Double-buffered LDS; async global->LDS staging pipelined against WMMA compute.
// Output f32 (C) or bf16 (Cb) selected by non-null pointer.
// Requires N % 128 == 0, K % 32 == 0. M edge handled (clamped loads, masked stores).
__global__ void __launch_bounds__(256) k_gemm(const unsigned short* __restrict__ A, int lda,
                                              const unsigned short* __restrict__ Bt, int ldb,
                                              float* __restrict__ C,
                                              unsigned short* __restrict__ Cb, int ldc,
                                              const float* __restrict__ bias,
                                              const float* __restrict__ resid,
                                              int M, int N, int K) {
    __shared__ __align__(16) unsigned short As[2][128 * 40];
    __shared__ __align__(16) unsigned short Bs[2][128 * 40];

    const int tid  = threadIdx.x;
    const int lane = tid & 31;
    const int wv   = tid >> 5;
    const int wm   = wv >> 2;      // 0..1
    const int wn   = wv & 3;       // 0..3
    const int bm0  = blockIdx.y * 128;
    const int bn0  = blockIdx.x * 128;

    const v8f z8 = {0.f,0.f,0.f,0.f,0.f,0.f,0.f,0.f};
    v8f acc[4][2];
    for (int i = 0; i < 4; ++i) for (int j = 0; j < 2; ++j) acc[i][j] = z8;

    // 128 rows x 32 k per tile, 2 threads per row, 16 bf16 (2 x b128) per thread.
    // A row clamped (not masked) so every wave issues exactly 4 async loads/tile:
    // required for the ASYNCcnt<=4 pipelined wait. Clamped rows only feed C rows
    // >= M, which are never stored.
    const int trow = tid >> 1;
    const int tc0  = (tid & 1) * 16;
    const int arow = (bm0 + trow < M) ? (bm0 + trow) : (M - 1);
    const unsigned short* Ap = A  + (size_t)arow * lda + tc0;
    const unsigned short* Bp = Bt + (size_t)(bn0 + trow) * ldb + tc0;
    const unsigned aoff[2] = { lds_off32(&As[0][trow * 40 + tc0]),
                               lds_off32(&As[1][trow * 40 + tc0]) };
    const unsigned boff[2] = { lds_off32(&Bs[0][trow * 40 + tc0]),
                               lds_off32(&Bs[1][trow * 40 + tc0]) };

    auto issue = [&](int buf, int kb) {
        async_ld_b128(aoff[buf],      Ap + kb);
        async_ld_b128(aoff[buf] + 16, Ap + kb + 8);
        async_ld_b128(boff[buf],      Bp + kb);
        async_ld_b128(boff[buf] + 16, Bp + kb + 8);
    };

    const int mlo = lane & 15;
    const int kba = (lane < 16) ? 0 : 8;    // 16-bit A layout: K 0-7/16-23 vs 8-15/24-31
    const int kbb = (lane < 16) ? 0 : 8;    // 16-bit B layout: K 0-7/16-23 vs 8-15/24-31

    const int ntiles = K / 32;
    issue(0, 0);
    for (int ib = 0; ib < ntiles; ++ib) {
        const int cur = ib & 1;
        __syncthreads();                      // all waves done reading buffer cur^1
        const bool more = (ib + 1 < ntiles);
        if (more) {
            issue(cur ^ 1, (ib + 1) * 32);    // next tile streams during compute
            if (ib + 2 < ntiles) {
                __builtin_prefetch(Ap + (ib + 2) * 32, 0, 1);
                __builtin_prefetch(Bp + (ib + 2) * 32, 0, 1);
            }
            wait_async4();                    // tile ib resident; tile ib+1 in flight
        } else {
            wait_async0();
        }
        __syncthreads();                      // whole tile visible to all waves

        Frag16 af[4];
        for (int mt = 0; mt < 4; ++mt) {
            int r = wm * 64 + mt * 16 + mlo;
            af[mt].u[0] = *(const uint4*)&As[cur][r * 40 + kba];
            af[mt].u[1] = *(const uint4*)&As[cur][r * 40 + kba + 16];
        }
        Frag16 bf[2];
        for (int nt = 0; nt < 2; ++nt) {
            int cidx = wn * 32 + nt * 16 + mlo;
            bf[nt].u[0] = *(const uint4*)&Bs[cur][cidx * 40 + kbb];
            bf[nt].u[1] = *(const uint4*)&Bs[cur][cidx * 40 + kbb + 16];
        }
        for (int mt = 0; mt < 4; ++mt)
            for (int nt = 0; nt < 2; ++nt)
                acc[mt][nt] = __builtin_amdgcn_wmma_f32_16x16x32_bf16(
                    false, af[mt].v, false, bf[nt].v, (short)0, acc[mt][nt], false, false);
    }

    const int rbase = (lane >> 4) * 8;
    for (int mt = 0; mt < 4; ++mt)
        for (int nt = 0; nt < 2; ++nt) {
            int gcol = bn0 + wn * 32 + nt * 16 + mlo;
            float bv = bias ? bias[gcol] : 0.f;
            for (int r = 0; r < 8; ++r) {
                int grow = bm0 + wm * 64 + mt * 16 + rbase + r;
                if (grow < M) {
                    float cv = acc[mt][nt][r] + bv;
                    if (resid) cv += resid[(size_t)grow * ldc + gcol];
                    if (Cb) Cb[(size_t)grow * ldc + gcol] = f2bf(cv);
                    else    C [(size_t)grow * ldc + gcol] = cv;
                }
            }
        }
}

// ---------------- fused flash attention (per b,h; 64 q-rows/block; WMMA) --------
// Q:(B,Nq,H,DH) bf16, K/V:(B,Nk,H,DH) bf16, O:(B,Nq,H,DH) bf16
// K tiles double-buffered via async global->LDS; V manual zero-filled transpose.
__global__ void __launch_bounds__(128) k_attn(const unsigned short* __restrict__ Q,
                                              const unsigned short* __restrict__ K,
                                              const unsigned short* __restrict__ V,
                                              unsigned short* __restrict__ O,
                                              int Nq, int Nk, float scale) {
    __shared__ __align__(16) unsigned short Kl[2][32 * 72];    // [key][d]
    __shared__ __align__(16) unsigned short Vt[64 * 40];       // [d][key]
    __shared__ __align__(16) unsigned short Pl[4 * 16 * 40];   // per-wave P transpose

    const int tid  = threadIdx.x;
    const int lane = tid & 31;
    const int wv   = tid >> 5;       // 0..3
    const int b    = blockIdx.x / H_;
    const int h    = blockIdx.x % H_;
    const int qb0  = blockIdx.y * 64;
    const int mlo  = lane & 15;
    const int hi   = lane >> 4;

    // Q fragments: four b128 loads (bf16, documented A-layout k-groups)
    Frag16 aq[2];
    {
        const int qrow = qb0 + wv * 16 + mlo;
        const size_t qbase = (size_t)(b * Nq + qrow) * (H_ * DH_) + (size_t)h * DH_;
        const int kbq = hi ? 8 : 0;
        for (int c = 0; c < 2; ++c) {
            aq[c].u[0] = *(const uint4*)(Q + qbase + c * 32 + kbq);
            aq[c].u[1] = *(const uint4*)(Q + qbase + c * 32 + kbq + 16);
        }
    }

    const v8f z8 = {0.f,0.f,0.f,0.f,0.f,0.f,0.f,0.f};
    v8f oacc[4];
    for (int t = 0; t < 4; ++t) oacc[t] = z8;
    float m8[8], l8[8];
    for (int r = 0; r < 8; ++r) { m8[r] = -1e30f; l8[r] = 0.f; }

    // per-thread staging assignment (32 keys x 64 d tile)
    const int key = tid >> 2;
    const int d0  = (tid & 3) * 16;
    const unsigned kloff[2] = { lds_off32(&Kl[0][key * 72 + d0]),
                                lds_off32(&Kl[1][key * 72 + d0]) };
    // K addresses clamped (uniform 2 async loads per wave per tile for the
    // ASYNCcnt<=2 pipelined wait); clamped key columns get masked to -1e30.
    auto issueK = [&](int buf, int kb0) {
        int kg = kb0 + key; if (kg >= Nk) kg = Nk - 1;
        const unsigned short* p = K + (size_t)(b * Nk + kg) * (H_ * DH_) + (size_t)h * DH_ + d0;
        async_ld_b128(kloff[buf],      p);
        async_ld_b128(kloff[buf] + 16, p + 8);
    };

    const int nkb = (Nk + 31) / 32;
    issueK(0, 0);
    for (int ib = 0; ib < nkb; ++ib) {
        const int cur = ib & 1;
        const int kb0 = ib * 32;
        __syncthreads();                    // prior reads of Kl[cur^1] and Vt done
        const bool more = (ib + 1 < nkb);
        if (more) issueK(cur ^ 1, kb0 + 32);
        // V tile: manual transposed store, zero-filled when invalid (0*NaN hazard in PV)
        {
            const int kg = kb0 + key;
            const bool ok = (kg < Nk);
            const unsigned short* vp = V + (size_t)(b * Nk + (ok ? kg : 0)) * (H_ * DH_)
                                         + (size_t)h * DH_ + d0;
            union { uint4 u; unsigned short s[8]; } v0c, v1c;
            v0c.u = make_uint4(0,0,0,0); v1c.u = v0c.u;
            if (ok) { v0c.u = *(const uint4*)vp; v1c.u = *(const uint4*)(vp + 8); }
            for (int j = 0; j < 8; ++j) {
                Vt[(d0 + j)     * 40 + key] = v0c.s[j];
                Vt[(d0 + 8 + j) * 40 + key] = v1c.s[j];
            }
        }
        if (more) wait_async2(); else wait_async0();
        __syncthreads();

        // S = Q * K^T (two 16x16 key tiles, d=64 contracted in two WMMA steps)
        v8f s[2];
        const int kbl = hi ? 8 : 0;
        for (int j = 0; j < 2; ++j) {
            const int keyn = j * 16 + mlo;
            Frag16 b0, b1;
            b0.u[0] = *(const uint4*)&Kl[cur][keyn * 72 + kbl];
            b0.u[1] = *(const uint4*)&Kl[cur][keyn * 72 + kbl + 16];
            b1.u[0] = *(const uint4*)&Kl[cur][keyn * 72 + 32 + kbl];
            b1.u[1] = *(const uint4*)&Kl[cur][keyn * 72 + 32 + kbl + 16];
            v8f sv = z8;
            sv = __builtin_amdgcn_wmma_f32_16x16x32_bf16(false, aq[0].v, false, b0.v, (short)0, sv, false, false);
            sv = __builtin_amdgcn_wmma_f32_16x16x32_bf16(false, aq[1].v, false, b1.v, (short)0, sv, false, false);
            s[j] = sv;
        }
        for (int j = 0; j < 2; ++j) {
            const bool valid = (kb0 + j * 16 + mlo) < Nk;
            for (int r = 0; r < 8; ++r) s[j][r] = valid ? s[j][r] * scale : -1e30f;
        }

        // online softmax (16-lane-group row reductions) + P transpose to LDS
        for (int r = 0; r < 8; ++r) {
            float s0 = s[0][r], s1 = s[1][r];
            float rmax = fmaxf(s0, s1);
            for (int off2 = 8; off2 >= 1; off2 >>= 1) rmax = fmaxf(rmax, __shfl_xor(rmax, off2, 32));
            float mn = fmaxf(m8[r], rmax);
            float p0 = __expf(s0 - mn), p1 = __expf(s1 - mn);
            float rs = p0 + p1;
            for (int off2 = 8; off2 >= 1; off2 >>= 1) rs += __shfl_xor(rs, off2, 32);
            float alpha = __expf(m8[r] - mn);
            l8[r] = l8[r] * alpha + rs;
            m8[r] = mn;
            for (int t = 0; t < 4; ++t) oacc[t][r] *= alpha;
            const int prow = r + 8 * hi;
            Pl[wv * 640 + prow * 40 + mlo]      = f2bf(p0);
            Pl[wv * 640 + prow * 40 + 16 + mlo] = f2bf(p1);
        }
        __syncthreads();

        // O += P * V
        Frag16 pf;
        const int kbp = hi ? 8 : 0;
        pf.u[0] = *(const uint4*)&Pl[wv * 640 + mlo * 40 + kbp];
        pf.u[1] = *(const uint4*)&Pl[wv * 640 + mlo * 40 + kbp + 16];
        const int kbv = hi ? 8 : 0;
        for (int t = 0; t < 4; ++t) {
            Frag16 vf;
            const int dn = t * 16 + mlo;
            vf.u[0] = *(const uint4*)&Vt[dn * 40 + kbv];
            vf.u[1] = *(const uint4*)&Vt[dn * 40 + kbv + 16];
            oacc[t] = __builtin_amdgcn_wmma_f32_16x16x32_bf16(false, pf.v, false, vf.v, (short)0, oacc[t], false, false);
        }
        // top-of-loop barrier protects Kl/Vt/Pl reuse
    }

    // write O (bf16, (B,Nq,H,DH) layout for the WO GEMM)
    for (int t = 0; t < 4; ++t)
        for (int r = 0; r < 8; ++r) {
            const int row = qb0 + wv * 16 + r + 8 * hi;
            const int col = t * 16 + mlo;
            const size_t oi = (size_t)(b * Nq + row) * (H_ * DH_) + (size_t)h * DH_ + col;
            O[oi] = f2bf(oacc[t][r] / l8[r]);
        }
}

// ================= host =================
extern "C" void kernel_launch(void* const* d_in, const int* in_sizes, int n_in,
                              void* d_out, int out_size, void* d_ws, size_t ws_size,
                              hipStream_t stream) {
    (void)in_sizes; (void)n_in; (void)out_size; (void)ws_size;
    char* ws = (char*)d_ws;
    size_t off = 0;
    auto alloc = [&](size_t bytes) -> char* {
        char* p = ws + off;
        off = (off + bytes + 255) & ~(size_t)255;
        return p;
    };

    // bf16 workspace buffers (all weights stored TRANSPOSED: Bt = W^T, [N][K])
    unsigned short* wq[3]; unsigned short* wk[3]; unsigned short* wv[3]; unsigned short* wo[3];
    for (int i = 0; i < 3; ++i) {
        wq[i] = (unsigned short*)alloc((size_t)D_ * INNER_ * 2);
        wk[i] = (unsigned short*)alloc((size_t)D_ * INNER_ * 2);
        wv[i] = (unsigned short*)alloc((size_t)D_ * INNER_ * 2);
        wo[i] = (unsigned short*)alloc((size_t)INNER_ * D_ * 2);
    }
    unsigned short* w1t    = (unsigned short*)alloc((size_t)D_ * 2 * FF_ * 2);
    unsigned short* w2t    = (unsigned short*)alloc((size_t)FF_ * D_ * 2);
    unsigned short* ctxbf  = (unsigned short*)alloc((size_t)B_ * CTX_ * D_ * 2);
    unsigned short* hintbf = (unsigned short*)alloc((size_t)B_ * TV_ * D_ * 2);
    unsigned short* xnbf   = (unsigned short*)alloc((size_t)B_ * N_ * D_ * 2);
    unsigned short* aobf   = (unsigned short*)alloc((size_t)B_ * N_ * INNER_ * 2);
    unsigned short* ggbf   = (unsigned short*)alloc((size_t)B_ * N_ * FF_ * 2);
    unsigned short* qb     = (unsigned short*)alloc((size_t)B_ * N_ * INNER_ * 2);
    unsigned short* kb     = (unsigned short*)alloc((size_t)B_ * N_ * INNER_ * 2);
    unsigned short* vb     = (unsigned short*)alloc((size_t)B_ * N_ * INNER_ * 2);
    float* xA = (float*)alloc((size_t)B_ * N_ * D_ * 4);
    float* xB = (float*)alloc((size_t)B_ * N_ * D_ * 4);
    float* pa = (float*)alloc((size_t)B_ * N_ * FF_ * 4);
    float* pg = (float*)alloc((size_t)B_ * N_ * FF_ * 4);

    // inputs
    const float* x     = (const float*)d_in[0];
    const float* ctx   = (const float*)d_in[1];
    const float* hint  = (const float*)d_in[2];
    const float* lng[4] = {(const float*)d_in[3], (const float*)d_in[5], (const float*)d_in[7], (const float*)d_in[9]};
    const float* lnb[4] = {(const float*)d_in[4], (const float*)d_in[6], (const float*)d_in[8], (const float*)d_in[10]};
    const float* fwq[3] = {(const float*)d_in[11], (const float*)d_in[16], (const float*)d_in[21]};
    const float* fwk[3] = {(const float*)d_in[12], (const float*)d_in[17], (const float*)d_in[22]};
    const float* fwv[3] = {(const float*)d_in[13], (const float*)d_in[18], (const float*)d_in[23]};
    const float* fwo[3] = {(const float*)d_in[14], (const float*)d_in[19], (const float*)d_in[24]};
    const float* fbo[3] = {(const float*)d_in[15], (const float*)d_in[20], (const float*)d_in[25]};
    const float* ffw1 = (const float*)d_in[26];
    const float* ffb1 = (const float*)d_in[27];
    const float* ffw2 = (const float*)d_in[28];
    const float* ffb2 = (const float*)d_in[29];

    auto cvt_t = [&](const float* s, unsigned short* d, int R, int C) {
        k_cvt_t<<<2048, 256, 0, stream>>>(s, d, R, C);
    };
    // weights -> transposed bf16 (WMMA-native precision; contiguous [n][k] tiles)
    for (int i = 0; i < 3; ++i) {
        cvt_t(fwq[i], wq[i], D_, INNER_);
        cvt_t(fwk[i], wk[i], D_, INNER_);
        cvt_t(fwv[i], wv[i], D_, INNER_);
        cvt_t(fwo[i], wo[i], INNER_, D_);
    }
    cvt_t(ffw1, w1t, D_, 2 * FF_);   // (2FF x D), halves at row offsets 0 / FF
    cvt_t(ffw2, w2t, FF_, D_);       // (D x FF)
    k_cvt<<<2048, 256, 0, stream>>>(ctx, ctxbf, (size_t)B_ * CTX_ * D_);
    k_hint_pe<<<2048, 256, 0, stream>>>(hint, hintbf, TV_, D_, (size_t)B_ * TV_ * D_);

    hipMemcpyAsync(xA, x, (size_t)B_ * N_ * D_ * 4, hipMemcpyDeviceToDevice, stream);

    auto gemm = [&](const unsigned short* A, int lda, const unsigned short* Bt, int ldb,
                    float* C, unsigned short* Cb, int ldc,
                    const float* bias, const float* resid,
                    int M, int N, int K) {
        dim3 g(N / 128, (M + 127) / 128);
        k_gemm<<<g, 256, 0, stream>>>(A, lda, Bt, ldb, C, Cb, ldc, bias, resid, M, N, K);
    };

    auto block = [&](const float* xin, float* xout, int lnIdx, int aIdx,
                     const unsigned short* kvsrc, int kvrows) {
        k_ln<<<B_ * N_, 256, 0, stream>>>(xin, lng[lnIdx], lnb[lnIdx], xnbf, D_);
        gemm(xnbf,  D_, wq[aIdx], D_, nullptr, qb, INNER_, nullptr, nullptr, B_ * N_,      INNER_, D_);
        gemm(kvsrc, D_, wk[aIdx], D_, nullptr, kb, INNER_, nullptr, nullptr, B_ * kvrows,  INNER_, D_);
        gemm(kvsrc, D_, wv[aIdx], D_, nullptr, vb, INNER_, nullptr, nullptr, B_ * kvrows,  INNER_, D_);
        dim3 ag(B_ * H_, N_ / 64);
        k_attn<<<ag, 128, 0, stream>>>(qb, kb, vb, aobf, N_, kvrows, 0.125f);
        gemm(aobf, INNER_, wo[aIdx], INNER_, xout, nullptr, D_, fbo[aIdx], xin, B_ * N_, D_, INNER_);
    };

    // attn1: self (kv = layernormed x); attn2: cross(context); attn3: cross(hint+PE), ln4
    block(xA, xB, 0, 0, xnbf, N_);
    block(xB, xA, 1, 1, ctxbf, CTX_);
    block(xA, xB, 3, 2, hintbf, TV_);

    // GEGLU feed-forward (ln3)
    k_ln<<<B_ * N_, 256, 0, stream>>>(xB, lng[2], lnb[2], xnbf, D_);
    gemm(xnbf, D_, w1t,                       D_, pa, nullptr, FF_, ffb1,       nullptr, B_ * N_, FF_, D_);
    gemm(xnbf, D_, w1t + (size_t)FF_ * D_,    D_, pg, nullptr, FF_, ffb1 + FF_, nullptr, B_ * N_, FF_, D_);
    k_geglu<<<4096, 256, 0, stream>>>(pa, pg, ggbf, (size_t)B_ * N_ * FF_);
    gemm(ggbf, FF_, w2t, FF_, (float*)d_out, nullptr, D_, ffb2, xB, B_ * N_, D_, FF_);
}
